// StackedLSTMDecoder_21114059227673
// MI455X (gfx1250) — compile-verified, matrix operand order checked by reading, not run on hardware
//
#include <hip/hip_runtime.h>
#include <hip/hip_bf16.h>

#define B 16
#define T 128
#define V 32000
#define H 512
#define L 2
#define D_DOCS 8
#define A_H 256
#define PAD_ID 0
#define EOS_ID 3
#define G4H 2048  /* 4*H */

typedef __attribute__((ext_vector_type(16))) __bf16 v16bf;
typedef __attribute__((ext_vector_type(8)))  float  v8f;
typedef unsigned int u32x4 __attribute__((ext_vector_type(4)));
typedef int i32x8 __attribute__((ext_vector_type(8)));
typedef int i32x4 __attribute__((ext_vector_type(4)));

// float -> bf16 round-to-nearest-even
static __device__ __forceinline__ __bf16 f2bf(float f) {
  union { float f; unsigned u; } v; v.f = f;
  unsigned r = v.u + 0x7FFFu + ((v.u >> 16) & 1u);
  unsigned short h = (unsigned short)(r >> 16);
  union { unsigned short s; __bf16 b; } o; o.s = h; return o.b;
}

// ISA 7.12.2: 16-bit A-matrix 16x32 element->K mapping (mirrored for B columns)
static __device__ __forceinline__ int kmap(int e, int lane) {
  return ((e >> 3) << 4) + ((lane >> 4) << 3) + (e & 7);
}

static __device__ __forceinline__ float sigmoidf(float x) { return 1.0f / (1.0f + __expf(-x)); }

// ---------------------------------------------------------------------------
// TDM: contiguous 16 KB global -> LDS copy via Tensor Data Mover.
// D# per ISA 08_async_tensor §8: 1-D tile, data_size=8B, tile_dim0=2048.
// Issue from one wave (EXEC ignored by TDM); caller does s_wait_tensorcnt+bar.
// This toolchain exposes the 6-arg builtin: (g0, g1, g2, g3, g_extra, cpol).
// ---------------------------------------------------------------------------
static __device__ __forceinline__ void tdm_load_16k(const void* gptr, void* lds_ptr) {
  unsigned long long ga = (unsigned long long)gptr;
  unsigned lds = (unsigned)(unsigned long long)lds_ptr;  // low 32b of LDS aperture = offset
  u32x4 g0;
  g0[0] = 1u;                                           // count=1 (valid user D#)
  g0[1] = lds;                                          // lds_addr (bytes)
  g0[2] = (unsigned)(ga & 0xFFFFFFFFu);                 // global_addr[31:0]
  g0[3] = (unsigned)((ga >> 32) & 0x1FFFFFFu) | (2u << 30);  // global_addr[56:32], type=2
  i32x8 g1;
  g1[0] = 0x00030000;                 // workgroup_mask=0, data_size=3 (8-byte elems)
  g1[1] = (int)(2048u << 16);         // tensor_dim0[15:0]=2048 (bits 63:48)
  g1[2] = (int)(1u << 16);            // tensor_dim0[31:16]=0, tensor_dim1=1
  g1[3] = (int)(2048u << 16);         // tile_dim0=2048 (bits 127:112)
  g1[4] = 0;                          // tile_dim1=0 (unused), tile_dim2=0
  g1[5] = 2048;                       // tensor_dim0_stride[31:0]
  g1[6] = 0;
  g1[7] = 0;
  i32x4 z4 = {0, 0, 0, 0};
  i32x8 z8 = {0, 0, 0, 0, 0, 0, 0, 0};
  __builtin_amdgcn_tensor_load_to_lds(g0, g1, z4, z4, z8, 0);
}

// ---------------------------------------------------------------------------
// Attention context precompute (tiny, VALU) + eos init. One block per (b,l).
// ---------------------------------------------------------------------------
__global__ void k_context(const float* __restrict__ embs,   // [B,D,L,H]
                          const float* __restrict__ W1,     // [A_H,H]
                          const float* __restrict__ b1,     // [A_H]
                          const float* __restrict__ W2,     // [1,A_H]
                          const float* __restrict__ b2,     // [1]
                          float* __restrict__ context,      // [B,L,H]
                          int* __restrict__ eos)            // [B]
{
  int b = blockIdx.x >> 1, l = blockIdx.x & 1, tid = threadIdx.x; // 256 threads
  __shared__ float red[A_H];
  __shared__ float wts[D_DOCS];
  float score[D_DOCS];
  for (int d = 0; d < D_DOCS; ++d) {
    const float* e = embs + (((size_t)(b * D_DOCS + d) * L + l) * H);
    const float* w = W1 + (size_t)tid * H;
    float acc = b1[tid];
    for (int k = 0; k < H; ++k) acc += e[k] * w[k];
    red[tid] = tanhf(acc) * W2[tid];
    __syncthreads();
    for (int s = A_H >> 1; s > 0; s >>= 1) {
      if (tid < s) red[tid] += red[tid + s];
      __syncthreads();
    }
    score[d] = red[0] + b2[0];
    __syncthreads();
  }
  if (tid == 0) {
    float m = score[0];
    for (int d = 1; d < D_DOCS; ++d) m = fmaxf(m, score[d]);
    float s = 0.f;
    for (int d = 0; d < D_DOCS; ++d) { wts[d] = __expf(score[d] - m); s += wts[d]; }
    float inv = 1.0f / s;
    for (int d = 0; d < D_DOCS; ++d) wts[d] *= inv;
  }
  __syncthreads();
  for (int h = tid; h < H; h += A_H) {
    float c = 0.f;
    for (int d = 0; d < D_DOCS; ++d)
      c += wts[d] * embs[(((size_t)(b * D_DOCS + d) * L + l) * H) + h];
    context[(size_t)(b * L + l) * H + h] = c;
  }
  if (blockIdx.x == 0 && tid < B) eos[tid] = 0;
}

// ---------------------------------------------------------------------------
// Pack a row-major f32 weight [Ntotal, H] into bf16 WMMA B-fragments.
// Fragment id = nt*16 + kt ; within fragment: lane*16 + e (contiguous 32B/lane)
// ---------------------------------------------------------------------------
__global__ void k_pack_b(const float* __restrict__ W, __bf16* __restrict__ dst, int total)
{
  for (int idx = blockIdx.x * blockDim.x + threadIdx.x; idx < total;
       idx += gridDim.x * blockDim.x) {
    int frag = idx >> 9, r = idx & 511, lane = r >> 4, e = r & 15;
    int kt = frag & 15, nt = frag >> 4;
    int n = (nt << 4) + (lane & 15);
    int k = (kt << 5) + kmap(e, lane);
    dst[idx] = f2bf(W[(size_t)n * H + k]);
  }
}

// ---------------------------------------------------------------------------
// Fused LSTM gate GEMM: gates[16,2048] = x@W_ih^T + hmix@W_hh^T + b_ih + b_hh.
// Phase 1: build A-fragments in LDS.
//   layer 0: x = embedding gather; layer 1: x = A_h0 frags streamed in by TDM.
//   hmix = 0.5*context + 0.5*hidden[layer], gathered + packed.
// Phase 2: 8 waves x 16 k-steps x 2 WMMAs, A from LDS, B from L2.
// 16 blocks x 256 threads.
// ---------------------------------------------------------------------------
__global__ void k_gates(const float* __restrict__ embed_W,
                        const int* __restrict__ init_input,
                        const int* __restrict__ targets,
                        const float* __restrict__ hidden,    // [B,L,H]
                        const float* __restrict__ context,   // [B,L,H]
                        const __bf16* __restrict__ Axf,      // layer1: packed h0 frags
                        const __bf16* __restrict__ Bih, const __bf16* __restrict__ Bhh,
                        const float* __restrict__ bih, const float* __restrict__ bhh,
                        float* __restrict__ gates, int layer, int t)
{
  __shared__ __align__(128) __bf16 aX[16 * 512];  // 16 KB
  __shared__ __align__(128) __bf16 aM[16 * 512];  // 16 KB
  int tid = threadIdx.x;
  int lane = tid & 31, wave = tid >> 5;

  if (layer == 1 && tid < 32) tdm_load_16k(Axf, aX);  // overlap with gathers below

  for (int idx = tid; idx < 16 * 512; idx += 256) {
    int kt = idx >> 9, r = idx & 511, ln = r >> 4, e = r & 15;
    int m = ln & 15;
    int k = (kt << 5) + kmap(e, ln);
    if (layer == 0) {
      int tok = (t == 0) ? init_input[m] : targets[m * T + (t - 1)];
      aX[idx] = f2bf(embed_W[(size_t)tok * H + k]);
    }
    float c = context[(m * L + layer) * H + k];
    float h = hidden[(m * L + layer) * H + k];
    aM[idx] = f2bf(0.5f * c + 0.5f * h);
  }
  if (layer == 1 && tid < 32) __builtin_amdgcn_s_wait_tensorcnt(0);
  __syncthreads();

  int nt = blockIdx.x * 8 + wave;
  v8f acc = {};
#pragma unroll
  for (int kt = 0; kt < 16; ++kt) {
    v16bf a1 = *(const v16bf*)(aX + kt * 512 + lane * 16);
    v16bf b1 = *(const v16bf*)(Bih + ((size_t)(nt * 16 + kt)) * 512 + lane * 16);
    acc = __builtin_amdgcn_wmma_f32_16x16x32_bf16(false, a1, false, b1, (short)0, acc, false, false);
    v16bf a2 = *(const v16bf*)(aM + kt * 512 + lane * 16);
    v16bf b2 = *(const v16bf*)(Bhh + ((size_t)(nt * 16 + kt)) * 512 + lane * 16);
    acc = __builtin_amdgcn_wmma_f32_16x16x32_bf16(false, a2, false, b2, (short)0, acc, false, false);
  }
  int n = (nt << 4) + (lane & 15);
  float bias = bih[n] + bhh[n];
  int mh = (lane >> 4) << 3;  // D-frag: lane holds N=lane&15, M = v + 8*(lane>>4)
#pragma unroll
  for (int v = 0; v < 8; ++v) gates[(mh + v) * G4H + n] = acc[v] + bias;
}

// ---------------------------------------------------------------------------
// LSTM cell activations; writes new hidden/cell state and packs h into
// A-fragments (LDS-staged) for the downstream TDM consumers. 1 block x 512.
// ---------------------------------------------------------------------------
__global__ void k_act(const float* __restrict__ gates, float* __restrict__ hidden,
                      float* __restrict__ cell, __bf16* __restrict__ Aout, int l)
{
  __shared__ float hsh[B * H];
  int tid = threadIdx.x;
  for (int u = tid; u < B * H; u += 512) {
    int b = u >> 9, j = u & 511;
    float gi = gates[b * G4H + j];
    float gf = gates[b * G4H + 512 + j];
    float gg = gates[b * G4H + 1024 + j];
    float go = gates[b * G4H + 1536 + j];
    float c = sigmoidf(gf) * cell[(b * L + l) * H + j] + sigmoidf(gi) * tanhf(gg);
    float h = sigmoidf(go) * tanhf(c);
    cell[(b * L + l) * H + j] = c;
    hidden[(b * L + l) * H + j] = h;
    hsh[u] = h;
  }
  __syncthreads();
  for (int idx = tid; idx < 16 * 512; idx += 512) {
    int kt = idx >> 9, r = idx & 511, lane = r >> 4, e = r & 15;
    int m = lane & 15;
    int k = (kt << 5) + kmap(e, lane);
    Aout[idx] = f2bf(hsh[m * H + k]);
  }
}

// ---------------------------------------------------------------------------
// Output projection: logits[16,32000] = h1@W_out^T + b_out.
// A-fragments arrive via TDM into LDS once per block; B fragments stream
// straight from L2 into VGPRs (no reuse -> no LDS staging).
// 250 blocks x 8 waves = 2000 N-tiles; 16 WMMAs per wave.
// ---------------------------------------------------------------------------
__global__ void k_logits(const __bf16* __restrict__ Ahf, const __bf16* __restrict__ Bw,
                         const float* __restrict__ bout, float* __restrict__ logits)
{
  __shared__ __align__(128) __bf16 aX[16 * 512];  // 16 KB
  int tid = threadIdx.x;
  int lane = tid & 31, wave = tid >> 5;
  if (tid < 32) {
    tdm_load_16k(Ahf, aX);
    __builtin_amdgcn_s_wait_tensorcnt(0);
  }
  __syncthreads();

  int nt = blockIdx.x * 8 + wave;
  v8f acc = {};
#pragma unroll
  for (int kt = 0; kt < 16; ++kt) {
    v16bf a = *(const v16bf*)(aX + kt * 512 + lane * 16);
    v16bf b = *(const v16bf*)(Bw + ((size_t)(nt * 16 + kt)) * 512 + lane * 16);
    acc = __builtin_amdgcn_wmma_f32_16x16x32_bf16(false, a, false, b, (short)0, acc, false, false);
  }
  int n = (nt << 4) + (lane & 15);
  float bias = bout[n];
  int mh = (lane >> 4) << 3;
#pragma unroll
  for (int v = 0; v < 8; ++v) logits[(size_t)(mh + v) * V + n] = acc[v] + bias;
}

// ---------------------------------------------------------------------------
// Softmax + argmax + EOS masking. One block per batch row; 3 passes over V.
// Writes probs into d_out[(b*T+t)*V + :] and id (as float) into tail region.
// ---------------------------------------------------------------------------
__global__ void k_softmax(const float* __restrict__ logits, int* __restrict__ eos,
                          float* __restrict__ out, int t)
{
  int b = blockIdx.x, tid = threadIdx.x;  // 256 threads
  __shared__ float smax[256];
  __shared__ int   sidx[256];
  __shared__ float ssum[256];
  __shared__ float s_max, s_sum;
  __shared__ int   s_arg, s_eosprev;
  const float* lg = logits + (size_t)b * V;

  float mv = -INFINITY; int mi = 0;
  for (int n = tid; n < V; n += 256) {
    float x = lg[n];
    if (x > mv || (x == mv && n < mi)) { mv = x; mi = n; }
  }
  smax[tid] = mv; sidx[tid] = mi; __syncthreads();
  for (int s = 128; s > 0; s >>= 1) {
    if (tid < s) {
      float ov = smax[tid + s]; int oi = sidx[tid + s];
      if (ov > smax[tid] || (ov == smax[tid] && oi < sidx[tid])) { smax[tid] = ov; sidx[tid] = oi; }
    }
    __syncthreads();
  }
  if (tid == 0) { s_max = smax[0]; s_arg = sidx[0]; s_eosprev = eos[b]; }
  __syncthreads();

  float M = s_max;
  float acc = 0.f;
  for (int n = tid; n < V; n += 256) acc += __expf(lg[n] - M);
  ssum[tid] = acc; __syncthreads();
  for (int s = 128; s > 0; s >>= 1) { if (tid < s) ssum[tid] += ssum[tid + s]; __syncthreads(); }
  if (tid == 0) s_sum = ssum[0];
  __syncthreads();

  float inv = 1.0f / s_sum;
  int ep = s_eosprev;
  float* po = out + ((size_t)b * T + t) * V;
  if (ep) {
    for (int n = tid; n < V; n += 256) po[n] = (n == PAD_ID) ? 1.0f : 0.0f;
  } else {
    for (int n = tid; n < V; n += 256) po[n] = __expf(lg[n] - M) * inv;
  }
  if (tid == 0) {
    int id = ep ? PAD_ID : s_arg;
    out[(size_t)B * T * V + (size_t)b * T + t] = (float)id;
    eos[b] = ep | ((id == EOS_ID) ? 1 : 0);
  }
}

// ---------------------------------------------------------------------------
extern "C" void kernel_launch(void* const* d_in, const int* in_sizes, int n_in,
                              void* d_out, int out_size, void* d_ws, size_t ws_size,
                              hipStream_t stream)
{
  (void)in_sizes; (void)n_in; (void)out_size; (void)ws_size;
  const float* init_hidden = (const float*)d_in[0];
  const float* init_cell   = (const float*)d_in[1];
  const int*   init_input  = (const int*)d_in[2];
  const int*   targets     = (const int*)d_in[3];
  const float* attend_embs = (const float*)d_in[4];
  const float* embed_W     = (const float*)d_in[5];
  const float* W_ih        = (const float*)d_in[6];
  const float* W_hh        = (const float*)d_in[7];
  const float* b_ih        = (const float*)d_in[8];
  const float* b_hh        = (const float*)d_in[9];
  const float* W_out       = (const float*)d_in[10];
  const float* b_out       = (const float*)d_in[11];
  const float* attn_W1     = (const float*)d_in[12];
  const float* attn_b1     = (const float*)d_in[13];
  const float* attn_W2     = (const float*)d_in[14];
  const float* attn_b2     = (const float*)d_in[15];

  char* ws = (char*)d_ws;
  size_t off = 0;
  auto alloc = [&](size_t bytes) -> void* {
    void* p = ws + off;
    off = (off + bytes + 255) & ~(size_t)255;
    return p;
  };
  __bf16* Bout   = (__bf16*)alloc((size_t)V * H * 2);        // 32.75 MB
  __bf16* Bih    = (__bf16*)alloc((size_t)L * G4H * H * 2);  // 4.2 MB
  __bf16* Bhh    = (__bf16*)alloc((size_t)L * G4H * H * 2);  // 4.2 MB
  float* logits  = (float*)alloc((size_t)B * V * 4);         // 2 MB
  float* gates   = (float*)alloc((size_t)B * G4H * 4);
  float* hidden  = (float*)alloc((size_t)B * L * H * 4);
  float* cell    = (float*)alloc((size_t)B * L * H * 4);
  float* context = (float*)alloc((size_t)B * L * H * 4);
  __bf16* A_h0   = (__bf16*)alloc((size_t)16 * 512 * 2);
  __bf16* A_h1   = (__bf16*)alloc((size_t)16 * 512 * 2);
  int* eos       = (int*)alloc((size_t)B * 4);

  // Mutable state starts from the (const) inputs each launch (graph-capture safe).
  (void)hipMemcpyAsync(hidden, init_hidden, (size_t)B * L * H * 4,
                       hipMemcpyDeviceToDevice, stream);
  (void)hipMemcpyAsync(cell, init_cell, (size_t)B * L * H * 4,
                       hipMemcpyDeviceToDevice, stream);

  // One-time (per launch) precompute.
  k_context<<<B * L, A_H, 0, stream>>>(attend_embs, attn_W1, attn_b1, attn_W2, attn_b2,
                                       context, eos);
  k_pack_b<<<2048, 256, 0, stream>>>(W_out, Bout, V * H);
  for (int l = 0; l < L; ++l) {
    k_pack_b<<<512, 256, 0, stream>>>(W_ih + (size_t)l * G4H * H, Bih + (size_t)l * G4H * H, G4H * H);
    k_pack_b<<<512, 256, 0, stream>>>(W_hh + (size_t)l * G4H * H, Bhh + (size_t)l * G4H * H, G4H * H);
  }

  float* out = (float*)d_out;
  for (int t = 0; t < T; ++t) {
    k_gates<<<16, 256, 0, stream>>>(embed_W, init_input, targets, hidden, context,
                                    (const __bf16*)nullptr, Bih, Bhh, b_ih, b_hh,
                                    gates, 0, t);
    k_act<<<1, 512, 0, stream>>>(gates, hidden, cell, A_h0, 0);
    k_gates<<<16, 256, 0, stream>>>(embed_W, init_input, targets, hidden, context,
                                    A_h0, Bih + (size_t)G4H * H, Bhh + (size_t)G4H * H,
                                    b_ih + G4H, b_hh + G4H, gates, 1, t);
    k_act<<<1, 512, 0, stream>>>(gates, hidden, cell, A_h1, 1);
    k_logits<<<250, 256, 0, stream>>>(A_h1, Bout, b_out, logits);
    k_softmax<<<B, 256, 0, stream>>>(logits, eos, out, t);
  }
}
